// CrossMultiHeadedAttention_27582279974942
// MI455X (gfx1250) — matrix operand order repaired and checked by reading, not compiled, not run on hardware
//
#include <hip/hip_runtime.h>
#include <hip/hip_bf16.h>
#include <stdint.h>

// ---------------------------------------------------------------------------
// CDNA5 (gfx1250) fused multi-head cross attention, bf16 WMMA everywhere.
// B=4, Lq=Lk=2048, EMBED=KDIM=VDIM=512, H=8, DHEAD=64.
// ---------------------------------------------------------------------------

typedef __attribute__((ext_vector_type(16))) __bf16 v16bf;
typedef __attribute__((ext_vector_type(8)))  __bf16 bf16x8;
typedef __attribute__((ext_vector_type(4)))  __bf16 bf16x4;
typedef __attribute__((ext_vector_type(8)))  float  v8f;
typedef __attribute__((ext_vector_type(4)))  float  f32x4;

#define EMBED 512
#define LSEQ  2048
#define BATCH 4
#define HEADS 8
#define DHEAD 64
#define MROWS (BATCH * LSEQ)   // 8192

// ---- async global->LDS copy (gfx1250 GLOBAL_LOAD_ASYNC_TO_LDS_B128) -------
// Generic pointers to LDS carry the LDS byte offset in their low 32 bits.
__device__ __forceinline__ void async_copy_b128(const void* gptr, void* lptr) {
  unsigned lds = (unsigned)(uintptr_t)lptr;
  asm volatile("global_load_async_to_lds_b128 %0, %1, off"
               :: "v"(lds), "v"(gptr) : "memory");
}
__device__ __forceinline__ void async_wait0() {
  asm volatile("s_wait_asynccnt 0x0" ::: "memory");
}

// ---- WMMA fragment loaders ------------------------------------------------
// A-matrix 16x32 bf16, ISA layout: lanes 0-15 hold row m=lane, K={0..7,16..23};
// lanes 16-31 hold row m=lane-16, K={8..15,24..31}.
__device__ __forceinline__ v16bf frag_a(const __bf16* lds, int row0, int col0,
                                        int ld, int lane) {
  const int m = lane & 15, g = lane >> 4;
  const __bf16* p = lds + (row0 + m) * ld + col0 + 8 * g;
  union { v16bf v; bf16x8 h[2]; } u;
  u.h[0] = *(const bf16x8*)(p);
  u.h[1] = *(const bf16x8*)(p + 16);
  return u.v;
}

// B-matrix 32x16 bf16 read from an LDS tile holding B^T row-major:
// ldsT[n*ld + k]; lane n holds column n, K={8g..8g+7, 16+8g..16+8g+7}.
__device__ __forceinline__ v16bf frag_bT(const __bf16* ldsT, int n0, int k0,
                                         int ld, int lane) {
  const int n = lane & 15, g = lane >> 4;
  const __bf16* p = ldsT + (n0 + n) * ld + k0 + 8 * g;
  union { v16bf v; bf16x8 h[2]; } u;
  u.h[0] = *(const bf16x8*)(p);
  u.h[1] = *(const bf16x8*)(p + 16);
  return u.v;
}

__device__ __forceinline__ v8f wmma_bf16(v16bf a, v16bf b, v8f c) {
  return __builtin_amdgcn_wmma_f32_16x16x32_bf16(false, a, false, b,
                                                 (short)0, c, false, false);
}

// ---------------------------------------------------------------------------
// Kernel 1: Y = X(fp32)[8192,512] @ W(fp32)[512,512] + bias -> bf16,
// written head-split as [b, h, l, 64].
// Block = 256 thr (8 waves), tile 128x64, BK=64, each wave 32x32 (2x2 WMMA).
// ---------------------------------------------------------------------------
__global__ void __launch_bounds__(256)
qkv_proj_kernel(const float* __restrict__ X, const float* __restrict__ W,
                const float* __restrict__ bias, __bf16* __restrict__ Y) {
  __shared__ __bf16 Xs[128 * 72] __attribute__((aligned(16)));
  __shared__ __bf16 WsT[64 * 72] __attribute__((aligned(16)));

  const int tid  = threadIdx.x;
  const int lane = tid & 31, wave = tid >> 5;
  const int wr = wave & 3, wc = wave >> 2;       // 4x2 wave grid
  const int m0 = blockIdx.y * 128, n0 = blockIdx.x * 64;
  const int g = lane >> 4, ln = lane & 15;

  v8f acc[2][2];
#pragma unroll
  for (int i = 0; i < 2; ++i)
#pragma unroll
    for (int j = 0; j < 2; ++j) acc[i][j] = (v8f){0, 0, 0, 0, 0, 0, 0, 0};

  for (int kt = 0; kt < EMBED; kt += 64) {
    __syncthreads();
    // X tile 128x64: f32x4 global loads, bf16x4 LDS stores.
    for (int i = tid; i < 128 * 16; i += 256) {
      int r = i >> 4, c4 = (i & 15) * 4;
      f32x4 x = *(const f32x4*)&X[(size_t)(m0 + r) * EMBED + kt + c4];
      bf16x4 o = { (__bf16)x[0], (__bf16)x[1], (__bf16)x[2], (__bf16)x[3] };
      *(bf16x4*)&Xs[r * 72 + c4] = o;
    }
    // W tile 64x64, stored transposed (scatter bf16 stores).
    for (int i = tid; i < 64 * 16; i += 256) {
      int r = i >> 4, c4 = (i & 15) * 4;                  // r = k, c4 = n
      f32x4 w = *(const f32x4*)&W[(size_t)(kt + r) * EMBED + n0 + c4];
#pragma unroll
      for (int j = 0; j < 4; ++j) WsT[(c4 + j) * 72 + r] = (__bf16)w[j];
    }
    __syncthreads();
#pragma unroll
    for (int ks = 0; ks < 64; ks += 32) {
      v16bf a0 = frag_a(Xs, wr * 32,      ks, 72, lane);
      v16bf a1 = frag_a(Xs, wr * 32 + 16, ks, 72, lane);
      v16bf b0 = frag_bT(WsT, wc * 32,      ks, 72, lane);
      v16bf b1 = frag_bT(WsT, wc * 32 + 16, ks, 72, lane);
      acc[0][0] = wmma_bf16(a0, b0, acc[0][0]);
      acc[0][1] = wmma_bf16(a0, b1, acc[0][1]);
      acc[1][0] = wmma_bf16(a1, b0, acc[1][0]);
      acc[1][1] = wmma_bf16(a1, b1, acc[1][1]);
    }
  }

#pragma unroll
  for (int tm = 0; tm < 2; ++tm)
#pragma unroll
    for (int tn = 0; tn < 2; ++tn) {
      int col = n0 + wc * 32 + tn * 16 + ln;
      float bv = bias[col];
      int hh = col >> 6, dd = col & 63;
#pragma unroll
      for (int r = 0; r < 8; ++r) {
        int row = m0 + wr * 32 + tm * 16 + r + 8 * g;
        int bb = row >> 11, l = row & 2047;
        size_t idx = ((size_t)((bb * HEADS + hh) * LSEQ + l)) * DHEAD + dd;
        Y[idx] = (__bf16)(acc[tm][tn][r] + bv);
      }
    }
}

// ---------------------------------------------------------------------------
// Kernel 2: flash attention. Block = (64 q-rows, head, batch), 128 thr
// (4 waves, 16 rows each). Online softmax, fp32 stats/accumulators.
// ---------------------------------------------------------------------------
__global__ void __launch_bounds__(128)
attn_kernel(const __bf16* __restrict__ Q, const __bf16* __restrict__ K,
            const __bf16* __restrict__ V, __bf16* __restrict__ C) {
  __shared__ __bf16 Qs[64 * 72] __attribute__((aligned(16)));
  __shared__ __bf16 Ks[64 * 72] __attribute__((aligned(16)));
  __shared__ __bf16 Vt[64 * 72] __attribute__((aligned(16)));  // transposed [d][l]
  __shared__ __bf16 Ps[64 * 72] __attribute__((aligned(16)));

  const int tid = threadIdx.x, lane = tid & 31, wave = tid >> 5;
  const int g = lane >> 4, ln = lane & 15;
  const int q0 = blockIdx.x * 64, h = blockIdx.y, bb = blockIdx.z;
  const int wm = wave * 16;                      // this wave's 16 rows
  const float scale = 0.125f;                    // 1/sqrt(64)
  const size_t head = (size_t)(bb * HEADS + h) * LSEQ;

  // Q tile: async global->LDS (64 rows x 64 cols = 512 b128 transfers).
  for (int i = tid; i < 64 * 8; i += 128) {
    int r = i >> 3, c0 = (i & 7) * 8;
    async_copy_b128(&Q[(head + q0 + r) * DHEAD + c0], &Qs[r * 72 + c0]);
  }

  v8f cacc[4];
#pragma unroll
  for (int dt = 0; dt < 4; ++dt) cacc[dt] = (v8f){0, 0, 0, 0, 0, 0, 0, 0};
  float mrow[8], lrow[8];
#pragma unroll
  for (int r = 0; r < 8; ++r) { mrow[r] = -1e30f; lrow[r] = 0.0f; }

  for (int kc = 0; kc < LSEQ; kc += 64) {
    __syncthreads();                              // prev iter done with Ks/Vt
    // K tile: async copy, row-major (== B^T layout for Q.K^T).
    for (int i = tid; i < 64 * 8; i += 128) {
      int r = i >> 3, c0 = (i & 7) * 8;
      async_copy_b128(&K[(head + kc + r) * DHEAD + c0], &Ks[r * 72 + c0]);
    }
    // V tile: vector load, transposed scatter into LDS.
    for (int i = tid; i < 64 * 8; i += 128) {
      int r = i >> 3, c0 = (i & 7) * 8;
      bf16x8 v = *(const bf16x8*)&V[(head + kc + r) * DHEAD + c0];
#pragma unroll
      for (int j = 0; j < 8; ++j) Vt[(c0 + j) * 72 + r] = v[j];
    }
    // Prefetch next chunk (one cache line per lane).
    if (kc + 64 < LSEQ) {
      int pr = tid >> 1, pc = (tid & 1) * 32;
      __builtin_prefetch(&K[(head + kc + 64 + pr) * DHEAD + pc], 0, 1);
      __builtin_prefetch(&V[(head + kc + 64 + pr) * DHEAD + pc], 0, 1);
    }
    async_wait0();                                // async writes visible
    __syncthreads();

    // S = Q * K^T  (16x64 per wave)
    v8f s[4];
#pragma unroll
    for (int ct = 0; ct < 4; ++ct) {
      v8f a = (v8f){0, 0, 0, 0, 0, 0, 0, 0};
#pragma unroll
      for (int d0 = 0; d0 < 64; d0 += 32) {
        v16bf fa = frag_a(Qs, wm, d0, 72, lane);
        v16bf fb = frag_bT(Ks, ct * 16, d0, 72, lane);
        a = wmma_bf16(fa, fb, a);
      }
      s[ct] = a;
    }

    // Online softmax. D-layout: VGPR r / lane -> row wm + r + 8*g, col ln.
#pragma unroll
    for (int r = 0; r < 8; ++r) {
      float smax = fmaxf(fmaxf(s[0][r], s[1][r]), fmaxf(s[2][r], s[3][r])) * scale;
#pragma unroll
      for (int mk = 8; mk >= 1; mk >>= 1)
        smax = fmaxf(smax, __shfl_xor(smax, mk, 32));
      float mnew  = fmaxf(mrow[r], smax);
      float alpha = __expf(mrow[r] - mnew);
      float rs = 0.0f;
#pragma unroll
      for (int ct = 0; ct < 4; ++ct) {
        float p = __expf(s[ct][r] * scale - mnew);
        s[ct][r] = p;
        rs += p;
      }
#pragma unroll
      for (int mk = 8; mk >= 1; mk >>= 1) rs += __shfl_xor(rs, mk, 32);
      lrow[r] = lrow[r] * alpha + rs;
      mrow[r] = mnew;
#pragma unroll
      for (int dt = 0; dt < 4; ++dt) cacc[dt][r] *= alpha;
    }

    // P -> LDS (bf16); each wave writes/reads only its own 16 rows.
#pragma unroll
    for (int ct = 0; ct < 4; ++ct)
#pragma unroll
      for (int r = 0; r < 8; ++r)
        Ps[(wm + r + 8 * g) * 72 + ct * 16 + ln] = (__bf16)s[ct][r];

    // ctx += P * V
#pragma unroll
    for (int dt = 0; dt < 4; ++dt)
#pragma unroll
      for (int j0 = 0; j0 < 64; j0 += 32) {
        v16bf fa = frag_a(Ps, wm, j0, 72, lane);
        v16bf fb = frag_bT(Vt, dt * 16, j0, 72, lane);
        cacc[dt] = wmma_bf16(fa, fb, cacc[dt]);
      }
  }

  // Normalize and store ctx as [b, l, h*64] bf16.
  float inv[8];
#pragma unroll
  for (int r = 0; r < 8; ++r) inv[r] = 1.0f / lrow[r];
#pragma unroll
  for (int dt = 0; dt < 4; ++dt)
#pragma unroll
    for (int r = 0; r < 8; ++r) {
      int row = q0 + wm + r + 8 * g;
      int col = h * DHEAD + dt * 16 + ln;
      C[((size_t)bb * LSEQ + row) * EMBED + col] = (__bf16)(cacc[dt][r] * inv[r]);
    }
}

// ---------------------------------------------------------------------------
// Kernel 3: out = ctx(bf16)[8192,512] @ Wo(fp32) + bo -> fp32 row-major.
// ---------------------------------------------------------------------------
__global__ void __launch_bounds__(256)
out_proj_kernel(const __bf16* __restrict__ Xc, const float* __restrict__ W,
                const float* __restrict__ bias, float* __restrict__ out) {
  __shared__ __bf16 Xs[128 * 72] __attribute__((aligned(16)));
  __shared__ __bf16 WsT[64 * 72] __attribute__((aligned(16)));

  const int tid  = threadIdx.x;
  const int lane = tid & 31, wave = tid >> 5;
  const int wr = wave & 3, wc = wave >> 2;
  const int m0 = blockIdx.y * 128, n0 = blockIdx.x * 64;
  const int g = lane >> 4, ln = lane & 15;

  v8f acc[2][2];
#pragma unroll
  for (int i = 0; i < 2; ++i)
#pragma unroll
    for (int j = 0; j < 2; ++j) acc[i][j] = (v8f){0, 0, 0, 0, 0, 0, 0, 0};

  for (int kt = 0; kt < EMBED; kt += 64) {
    __syncthreads();
    // X tile (already bf16): async b128 global->LDS copies.
    for (int i = tid; i < 128 * 8; i += 256) {
      int r = i >> 3, c8 = (i & 7) * 8;
      async_copy_b128(&Xc[(size_t)(m0 + r) * EMBED + kt + c8], &Xs[r * 72 + c8]);
    }
    // W tile 64x64, transposed.
    for (int i = tid; i < 64 * 16; i += 256) {
      int r = i >> 4, c4 = (i & 15) * 4;
      f32x4 w = *(const f32x4*)&W[(size_t)(kt + r) * EMBED + n0 + c4];
#pragma unroll
      for (int j = 0; j < 4; ++j) WsT[(c4 + j) * 72 + r] = (__bf16)w[j];
    }
    async_wait0();
    __syncthreads();
#pragma unroll
    for (int ks = 0; ks < 64; ks += 32) {
      v16bf a0 = frag_a(Xs, wr * 32,      ks, 72, lane);
      v16bf a1 = frag_a(Xs, wr * 32 + 16, ks, 72, lane);
      v16bf b0 = frag_bT(WsT, wc * 32,      ks, 72, lane);
      v16bf b1 = frag_bT(WsT, wc * 32 + 16, ks, 72, lane);
      acc[0][0] = wmma_bf16(a0, b0, acc[0][0]);
      acc[0][1] = wmma_bf16(a0, b1, acc[0][1]);
      acc[1][0] = wmma_bf16(a1, b0, acc[1][0]);
      acc[1][1] = wmma_bf16(a1, b1, acc[1][1]);
    }
  }

#pragma unroll
  for (int tm = 0; tm < 2; ++tm)
#pragma unroll
    for (int tn = 0; tn < 2; ++tn) {
      int col = n0 + wc * 32 + tn * 16 + ln;
      float bv = bias[col];
#pragma unroll
      for (int r = 0; r < 8; ++r) {
        int row = m0 + wr * 32 + tm * 16 + r + 8 * g;
        out[(size_t)row * EMBED + col] = acc[tm][tn][r] + bv;
      }
    }
}

// ---------------------------------------------------------------------------
extern "C" void kernel_launch(void* const* d_in, const int* in_sizes, int n_in,
                              void* d_out, int out_size, void* d_ws, size_t ws_size,
                              hipStream_t stream) {
  (void)in_sizes; (void)n_in; (void)out_size; (void)ws_size;
  const float* query = (const float*)d_in[0];
  const float* key   = (const float*)d_in[1];
  const float* value = (const float*)d_in[2];
  const float* wq = (const float*)d_in[3];
  const float* bq = (const float*)d_in[4];
  const float* wk = (const float*)d_in[5];
  const float* bk = (const float*)d_in[6];
  const float* wv = (const float*)d_in[7];
  const float* bv = (const float*)d_in[8];
  const float* wo = (const float*)d_in[9];
  const float* bo = (const float*)d_in[10];

  // Workspace: 4 bf16 buffers of 8192*512 elements (8 MB each, 32 MB total).
  const size_t bufBytes = (size_t)MROWS * EMBED * sizeof(unsigned short);
  char* ws = (char*)d_ws;
  __bf16* Qw = (__bf16*)(ws + 0 * bufBytes);
  __bf16* Kw = (__bf16*)(ws + 1 * bufBytes);
  __bf16* Vw = (__bf16*)(ws + 2 * bufBytes);
  __bf16* Cw = (__bf16*)(ws + 3 * bufBytes);

  dim3 gproj(EMBED / 64, MROWS / 128);           // (8, 64)
  qkv_proj_kernel<<<gproj, 256, 0, stream>>>(query, wq, bq, Qw);
  qkv_proj_kernel<<<gproj, 256, 0, stream>>>(key,   wk, bk, Kw);
  qkv_proj_kernel<<<gproj, 256, 0, stream>>>(value, wv, bv, Vw);

  attn_kernel<<<dim3(LSEQ / 64, HEADS, BATCH), 128, 0, stream>>>(Qw, Kw, Vw, Cw);

  out_proj_kernel<<<gproj, 256, 0, stream>>>(Cw, wo, bo, (float*)d_out);
}
